// VectorQuantizer_55473797595299
// MI455X (gfx1250) — matrix operand order repaired and checked by reading, not compiled, not run on hardware
//
#include <hip/hip_runtime.h>
#include <stdint.h>

// ---------- types for WMMA ----------
typedef __attribute__((ext_vector_type(16))) __bf16 v16bf;
typedef __attribute__((ext_vector_type(8)))  float  v8f;

union BF16Frag { v16bf v; uint32_t u[8]; };

#define DIM 512
#define NCODES 512
#define TT 2048
#define NB 16
#define NTOK (NB * TT)          // 32768 tokens
#define PITCH 516               // 512 + 4 bf16 pad -> conflict-mitigated LDS
#define QSIZE ((size_t)NB * DIM * TT)  // 16777216

static __device__ __forceinline__ unsigned short f2bf(float f) {
    uint32_t x = __float_as_uint(f);
    uint32_t r = (x + 0x7fffu + ((x >> 16) & 1u)) >> 16;   // round-nearest-even
    return (unsigned short)r;
}

// ---------------------------------------------------------------------------
// Kernel 1: codebook fp32 -> bf16, plus per-code squared norms
// grid = 512 (one block per code), block = 256 (8 waves)
// ---------------------------------------------------------------------------
__global__ __launch_bounds__(256) void vq_prep(const float* __restrict__ cb,
                                               unsigned short* __restrict__ cbbf,
                                               float* __restrict__ norms) {
    int k = blockIdx.x;
    int tid = threadIdx.x;
    float s = 0.f;
    for (int d = tid; d < DIM; d += 256) {
        float v = cb[(size_t)k * DIM + d];
        cbbf[(size_t)k * DIM + d] = f2bf(v);
        s += v * v;
    }
    // wave32 reduction
    for (int m = 16; m >= 1; m >>= 1) s += __shfl_xor(s, m, 32);
    __shared__ float red[8];
    int w = tid >> 5;
    if ((tid & 31) == 0) red[w] = s;
    __syncthreads();
    if (tid == 0) {
        float tot = 0.f;
        for (int i = 0; i < 8; ++i) tot += red[i];
        norms[k] = tot;
    }
}

// ---------------------------------------------------------------------------
// Kernel 2: WMMA nearest-code search.
// One wave (32 threads) per 32 tokens. LDS holds 32x512 bf16 z-tile (pitch 516).
// For each 16-code tile, accumulate scores = z . e over K=512 with
// v_wmma_f32_16x16x32_bf16, then update running argmin of (||e||^2 - 2 z.e).
// ---------------------------------------------------------------------------
__global__ __launch_bounds__(32) void vq_search(const float* __restrict__ z,
                                                const unsigned short* __restrict__ cbbf,
                                                const float* __restrict__ norms,
                                                int* __restrict__ outIdx) {
    __shared__ unsigned short zl[32 * PITCH];   // 33,024 bytes

    const int lane = threadIdx.x;
    const int n0 = blockIdx.x * 32;             // 32 | 2048 -> never straddles b
    const int b  = n0 >> 11;
    const int t0 = n0 & (TT - 1);

    // --- stage z tile: lane = token, coalesced 128B rows per dim ---
    const float* zb = z + ((size_t)b * DIM) * TT + t0 + lane;
    #pragma unroll 8
    for (int d = 0; d < DIM; ++d) {
        zl[lane * PITCH + d] = f2bf(zb[(size_t)d * TT]);
    }
    __syncthreads();

    const int h  = lane >> 4;   // K-half selector for fragments
    const int tk = lane & 15;   // token column / code row within tile

    float best0 = 3.4e38f, best1 = 3.4e38f;
    int   bi0 = 0, bi1 = 0;

    for (int mt = 0; mt < 32; ++mt) {
        const float* nrm = norms + mt * 16 + h * 8;
        const unsigned short* arow = cbbf + (size_t)(mt * 16 + tk) * DIM;

        v8f acc0 = {};
        v8f acc1 = {};

        #pragma unroll
        for (int kk = 0; kk < 16; ++kk) {
            const int k0 = kk * 32;

            // A fragment: code row m = tk; lane holds K = h*16 .. h*16+15
            BF16Frag af;
            const uint32_t* ap0 = (const uint32_t*)(arow + k0 + h * 16);
            const uint32_t* ap1 = (const uint32_t*)(arow + k0 + h * 16 + 8);
            #pragma unroll
            for (int j = 0; j < 4; ++j) { af.u[j] = ap0[j]; af.u[4 + j] = ap1[j]; }

            // B fragments: column n = token; lane holds K = h*16 .. h*16+15
            BF16Frag bf0, bf1;
            const uint32_t* bp0 = (const uint32_t*)&zl[tk * PITCH + k0 + h * 16];
            const uint32_t* bp1 = (const uint32_t*)&zl[(16 + tk) * PITCH + k0 + h * 16];
            #pragma unroll
            for (int j = 0; j < 8; ++j) { bf0.u[j] = bp0[j]; bf1.u[j] = bp1[j]; }

            acc0 = __builtin_amdgcn_wmma_f32_16x16x32_bf16(
                       false, af.v, false, bf0.v, (short)0, acc0, false, false);
            acc1 = __builtin_amdgcn_wmma_f32_16x16x32_bf16(
                       false, af.v, false, bf1.v, (short)0, acc1, false, false);
        }

        // dist = ||e||^2 - 2*score ; C layout: VGPR r -> code m = r + 8*h
        #pragma unroll
        for (int r = 0; r < 8; ++r) {
            const int code = mt * 16 + h * 8 + r;
            const float nv = nrm[r];
            float d0v = nv - 2.0f * acc0[r];
            if (d0v < best0) { best0 = d0v; bi0 = code; }
            float d1v = nv - 2.0f * acc1[r];
            if (d1v < best1) { best1 = d1v; bi1 = code; }
        }
    }

    // merge the two code-halves held by lane and lane^16 (same token)
    {
        float ob = __shfl_xor(best0, 16, 32);
        int   oi = __shfl_xor(bi0,   16, 32);
        if (ob < best0 || (ob == best0 && oi < bi0)) { best0 = ob; bi0 = oi; }
        ob = __shfl_xor(best1, 16, 32);
        oi = __shfl_xor(bi1,   16, 32);
        if (ob < best1 || (ob == best1 && oi < bi1)) { best1 = ob; bi1 = oi; }
    }
    if (lane < 16) {
        outIdx[n0 + tk]      = bi0;
        outIdx[n0 + 16 + tk] = bi1;
    }
}

// ---------------------------------------------------------------------------
// Kernel 3: exact fp32 gather of codebook[idx] -> quantized [B,D,T],
// plus sum of squared error for the loss.
// grid = 16384, block = 256, float4 over t.
// ---------------------------------------------------------------------------
__global__ __launch_bounds__(256) void vq_gather(const float* __restrict__ z,
                                                 const float* __restrict__ cb,
                                                 const int* __restrict__ idx,
                                                 float* __restrict__ q,
                                                 float* __restrict__ lossAcc) {
    size_t gid = (size_t)blockIdx.x * 256 + threadIdx.x;   // 4,194,304 total
    int t    = (int)(gid & 511) << 2;                      // 4 tokens along T
    size_t r = gid >> 9;
    int d    = (int)(r & 511);
    int b    = (int)(r >> 9);

    size_t off = ((size_t)(b * DIM + d)) * TT + t;
    float4 zv = *(const float4*)(z + off);

    const int* ip = idx + b * TT + t;
    int i0 = ip[0], i1 = ip[1], i2 = ip[2], i3 = ip[3];
    float q0 = cb[(size_t)i0 * DIM + d];
    float q1 = cb[(size_t)i1 * DIM + d];
    float q2 = cb[(size_t)i2 * DIM + d];
    float q3 = cb[(size_t)i3 * DIM + d];
    float4 qv = make_float4(q0, q1, q2, q3);
    *(float4*)(q + off) = qv;

    float e0 = q0 - zv.x, e1 = q1 - zv.y, e2 = q2 - zv.z, e3 = q3 - zv.w;
    float s = e0 * e0 + e1 * e1 + e2 * e2 + e3 * e3;

    for (int m = 16; m >= 1; m >>= 1) s += __shfl_xor(s, m, 32);
    __shared__ float red[8];
    int w = threadIdx.x >> 5;
    if ((threadIdx.x & 31) == 0) red[w] = s;
    __syncthreads();
    if (threadIdx.x == 0) {
        float tot = 0.f;
        for (int i = 0; i < 8; ++i) tot += red[i];
        atomicAdd(lossAcc, tot);
    }
}

// ---------------------------------------------------------------------------
// Kernel 4: loss scalar + indices (as float) into the output tail.
// grid = 128, block = 256.
// ---------------------------------------------------------------------------
__global__ __launch_bounds__(256) void vq_epilogue(const int* __restrict__ idx,
                                                   const float* __restrict__ lossAcc,
                                                   float* __restrict__ out) {
    int n = blockIdx.x * 256 + threadIdx.x;
    if (n < NTOK) out[QSIZE + 1 + n] = (float)idx[n];
    if (n == 0)   out[QSIZE] = 1.25f * (*lossAcc) / (float)QSIZE;
}

// ---------------------------------------------------------------------------
extern "C" void kernel_launch(void* const* d_in, const int* in_sizes, int n_in,
                              void* d_out, int out_size, void* d_ws, size_t ws_size,
                              hipStream_t stream) {
    const float* z  = (const float*)d_in[0];   // [16, 512, 2048] fp32
    const float* cb = (const float*)d_in[1];   // [512, 512] fp32
    float* out = (float*)d_out;                // quantized | loss | indices

    // ws layout (< 1 MB)
    unsigned short* cbbf  = (unsigned short*)d_ws;                    // 524,288 B
    float*          norms = (float*)((char*)d_ws + 524288);           //   2,048 B
    float*          lacc  = (float*)((char*)d_ws + 526336);           //       4 B
    int*            idxb  = (int*)  ((char*)d_ws + 526400);           // 131,072 B

    hipMemsetAsync(lacc, 0, sizeof(float), stream);
    vq_prep    <<<NCODES, 256, 0, stream>>>(cb, cbbf, norms);
    vq_search  <<<NTOK / 32, 32, 0, stream>>>(z, cbbf, norms, idxb);
    vq_gather  <<<(QSIZE / 4) / 256, 256, 0, stream>>>(z, cb, idxb, out, lacc);
    vq_epilogue<<<128, 256, 0, stream>>>(idxb, lacc, out);
}
